// Potential_11828339933353
// MI455X (gfx1250) — matrix operand order repaired and checked by reading, not compiled, not run on hardware
//
#include <hip/hip_runtime.h>
#include <hip/hip_bf16.h>

// ---------------------------------------------------------------------------
// EGNN potential on MI455X (gfx1250): wave32 + v_wmma_f32_16x16x32_f16.
// Each source node owns 31 contiguous edges -> all segment reductions are
// done in LDS / in-wave (no atomics, deterministic).
// A-fragments are preloaded to registers once per GEMM; inner loops are pure
// (B-load clause -> wmma chain); biases staged in LDS.
// ---------------------------------------------------------------------------

typedef __attribute__((ext_vector_type(16))) _Float16 v16h;
typedef __attribute__((ext_vector_type(8)))  _Float16 v8h;
typedef __attribute__((ext_vector_type(8)))  float    v8f;

#define Gn   512
#define An   32
#define Nn   (Gn*An)      // 16384 nodes
#define HIDc 256
#define Lc   4
#define KP1  544          // 513 padded to 17 chunks of 32

// silu with fast v_rcp_f32 (no IEEE-division expansion)
__device__ __forceinline__ float silu_f(float x) {
  float e = __expf(-x);
  return x * __builtin_amdgcn_rcpf(1.0f + e);
}

// A-fragment for v_wmma_f32_16x16x32_f16 from a row-major f16 buffer.
// Lane l<16: row=l, K = kbase+{0..15}; lane>=16: row=l-16,
// K = kbase+{16..31}.  Caller passes p = rowptr + kbase + (lane<16?0:16).
__device__ __forceinline__ v16h ldsA_pair(const _Float16* p) {
  v8h lo = *(const v8h*)(p);
  v8h hi = *(const v8h*)(p + 8);
  v16h r;
#pragma unroll
  for (int i = 0; i < 8; ++i) { r[i] = lo[i]; r[i + 8] = hi[i]; }
  return r;
}

// B-fragment: weights stored transposed+f16 as wt[n][kpad] (contiguous K per
// output column).  Lane l: col = l&15, K = kbase + (l<16 ? 0..15 : 16..31).
__device__ __forceinline__ v16h loadB(const _Float16* __restrict__ wt, int kpad,
                                      int ntile, int kbase, int lane) {
  int col = lane & 15;
  int ko  = (lane < 16) ? 0 : 16;
  return *(const v16h*)(wt + (size_t)(ntile * 16 + col) * kpad + kbase + ko);
}

__device__ __forceinline__ v8f wmma_f16(v16h a, v16h b, v8f c) {
  return __builtin_amdgcn_wmma_f32_16x16x32_f16(false, a, false, b, (short)0, c,
                                                false, false);
}

// dot of one LDS f16 row (256 wide) with an f32 vector in global memory
__device__ __forceinline__ float row_dot256(const _Float16* row,
                                            const float* __restrict__ w,
                                            float acc) {
  const v8h* r8 = (const v8h*)row;
  for (int c8 = 0; c8 < 32; ++c8) {
    v8h v = r8[c8];
    const float* wp = w + c8 * 8;
#pragma unroll
    for (int i = 0; i < 8; ++i) acc += (float)v[i] * wp[i];
  }
  return acc;
}

// ---------------------------------------------------------------------------
// Weight transpose + f32->f16 convert:  out[n*Kpad + k] = (k<K) ? w[k*N+n] : 0
// ---------------------------------------------------------------------------
__global__ void __launch_bounds__(256)
cvt_kernel(const float* __restrict__ w, _Float16* __restrict__ o,
           int K, int N, int Kpad) {
  int idx = blockIdx.x * 256 + threadIdx.x;
  if (idx >= N * Kpad) return;
  int n = idx / Kpad, k = idx - n * Kpad;
  o[idx] = (k < K) ? (_Float16)w[(size_t)k * N + n] : (_Float16)0.0f;
}

// ---------------------------------------------------------------------------
// remove_mean_batch: one wave per graph (32 lanes = 32 nodes)
// ---------------------------------------------------------------------------
__global__ void __launch_bounds__(64)
centroid_kernel(const float* __restrict__ pos, float* __restrict__ x0,
                float* __restrict__ xc) {
  int gid  = blockIdx.x * 2 + (threadIdx.x >> 5);
  int lane = threadIdx.x & 31;
  int n    = gid * 32 + lane;
  float p[3];
#pragma unroll
  for (int c = 0; c < 3; ++c) p[c] = pos[n * 3 + c];
#pragma unroll
  for (int c = 0; c < 3; ++c) {
    float v = p[c];
    for (int m = 16; m > 0; m >>= 1) v += __shfl_xor(v, m, 32);
    float r = p[c] - v * (1.0f / 32.0f);
    x0[n * 3 + c] = r;
    xc[n * 3 + c] = r;
  }
}

// ---------------------------------------------------------------------------
// Fragment encoder 11 -> 22 -> 255 (+time channel).  Scalar VALU, tiny.
// ---------------------------------------------------------------------------
__global__ void __launch_bounds__(256)
encoder_kernel(const float* __restrict__ nf,
               const float* __restrict__ w1, const float* __restrict__ b1,
               const float* __restrict__ w2, const float* __restrict__ b2,
               const float* __restrict__ tptr,
               float* __restrict__ h32, _Float16* __restrict__ h16) {
  int n = blockIdx.x * 256 + threadIdx.x;
  if (n >= Nn) return;
  float f[11];
#pragma unroll
  for (int i = 0; i < 11; ++i) f[i] = nf[n * 11 + i];
  float hid[22];
#pragma unroll
  for (int j = 0; j < 22; ++j) {
    float a = b1[j];
#pragma unroll
    for (int i = 0; i < 11; ++i) a += f[i] * w1[i * 22 + j];
    hid[j] = silu_f(a);
  }
  for (int o = 0; o < 255; ++o) {
    float a = b2[o];
#pragma unroll
    for (int j = 0; j < 22; ++j) a += hid[j] * w2[j * 255 + o];
    h32[(size_t)n * HIDc + o] = a;
    h16[(size_t)n * HIDc + o] = (_Float16)a;
  }
  float tv = tptr[0];
  h32[(size_t)n * HIDc + 255] = tv;
  h16[(size_t)n * HIDc + 255] = (_Float16)tv;
}

// ---------------------------------------------------------------------------
// Edge kernel: one block (64 thr = 2 waves) per SOURCE node.  31 edges
// (+1 pad row) -> two 16-row WMMA tiles.  phi_e, phi_x, agg, coord update.
// LDS: hi[256]h | hj[32][256]h | t0[32][256]h | t1[32][256]h |
//      rel[32][4]f | d2[32]f | cf[32]f | b1[256]f | b2[256]f | b3[256]f
// ---------------------------------------------------------------------------
#define EDGE_LDS 53504

__global__ void __launch_bounds__(64)
edge_kernel(const _Float16* __restrict__ h16,
            const float* __restrict__ xc, float* __restrict__ xn,
            float* __restrict__ agg,
            const _Float16* __restrict__ ew1t,   // [256][544]
            const _Float16* __restrict__ ew2t,   // [256][256]
            const _Float16* __restrict__ xw1t,   // [256][256]
            const float* __restrict__ eb1, const float* __restrict__ eb2,
            const float* __restrict__ xb1,
            const float* __restrict__ xw2, const float* __restrict__ xb2) {
  extern __shared__ char smem[];
  _Float16* s_hi  = (_Float16*)smem;
  _Float16* s_hj  = s_hi + 256;
  _Float16* s_t0  = s_hj + 32 * 256;
  _Float16* s_t1  = s_t0 + 32 * 256;
  float*    s_rel = (float*)(s_t1 + 32 * 256);
  float*    s_d2  = s_rel + 32 * 4;
  float*    s_cf  = s_d2 + 32;
  float*    s_b1  = s_cf + 32;
  float*    s_b2  = s_b1 + 256;
  float*    s_b3  = s_b2 + 256;

  const int tid  = threadIdx.x;
  const int lane = tid & 31;
  const int wave = tid >> 5;
  const int src  = blockIdx.x;
  const int g    = src >> 5;
  const int sl   = src & 31;

  // ---- stage inputs + biases ---------------------------------------------
  if (tid < 32)
    *(v8h*)(s_hi + tid * 8) = *(const v8h*)(h16 + (size_t)src * HIDc + tid * 8);
  for (int idx = tid; idx < 32 * 32; idx += 64) {
    int e = idx >> 5, ch = idx & 31;
    int dl  = e + (e >= sl ? 1 : 0);            // analytic dst (e==31 -> pad)
    int dst = (dl < 32) ? (g * 32 + dl) : src;
    *(v8h*)(s_hj + e * HIDc + ch * 8) =
        *(const v8h*)(h16 + (size_t)dst * HIDc + ch * 8);
  }
  if (tid < 32) {
    int e   = tid;
    int dl  = e + (e >= sl ? 1 : 0);
    int dst = (dl < 32) ? (g * 32 + dl) : src;
    float rx = xc[src * 3 + 0] - xc[dst * 3 + 0];
    float ry = xc[src * 3 + 1] - xc[dst * 3 + 1];
    float rz = xc[src * 3 + 2] - xc[dst * 3 + 2];
    s_rel[e * 4 + 0] = rx; s_rel[e * 4 + 1] = ry; s_rel[e * 4 + 2] = rz;
    s_d2[e] = rx * rx + ry * ry + rz * rz;
  }
  for (int i = tid; i < 256; i += 64) {
    s_b1[i] = eb1[i]; s_b2[i] = eb2[i]; s_b3[i] = xb1[i];
  }
  __syncthreads();

  const int r0    = wave * 16;
  const int l15   = lane & 15;
  const int myrow = r0 + l15;
  const int ko8   = (lane < 16) ? 0 : 16;
  const int mhi   = (lane < 16) ? 0 : 8;

  // ---- t0 = silu(e_in @ ew1 + eb1),  K = 544 (hi|hj|d2|pad) ---------------
  {
    v16h a1[17];
#pragma unroll
    for (int kc = 0; kc < 8; ++kc)
      a1[kc] = ldsA_pair(s_hi + kc * 32 + ko8);                 // h[src]
#pragma unroll
    for (int kc = 0; kc < 8; ++kc)
      a1[8 + kc] = ldsA_pair(s_hj + myrow * HIDc + kc * 32 + ko8); // h[dst]
    {
      v16h a;
#pragma unroll
      for (int i = 0; i < 16; ++i) a[i] = (_Float16)0.0f;
      if (lane < 16) a[0] = (_Float16)s_d2[myrow];              // d2 @ K=512
      a1[16] = a;
    }
    for (int nt = 0; nt < 16; ++nt) {
      float bia = s_b1[nt * 16 + l15];
      v8f c;
#pragma unroll
      for (int i = 0; i < 8; ++i) c[i] = bia;
#pragma unroll
      for (int kc = 0; kc < 17; ++kc) {
        v16h b = loadB(ew1t, KP1, nt, kc * 32, lane);
        c = wmma_f16(a1[kc], b, c);
      }
#pragma unroll
      for (int i = 0; i < 8; ++i)
        s_t0[(r0 + i + mhi) * HIDc + nt * 16 + l15] = (_Float16)silu_f(c[i]);
    }
  }

  // ---- m = silu(t0 @ ew2 + eb2) -> s_t1 -----------------------------------
  {
    v16h a2[8];
#pragma unroll
    for (int kc = 0; kc < 8; ++kc)
      a2[kc] = ldsA_pair(s_t0 + myrow * HIDc + kc * 32 + ko8);
    for (int nt = 0; nt < 16; ++nt) {
      float bia = s_b2[nt * 16 + l15];
      v8f c;
#pragma unroll
      for (int i = 0; i < 8; ++i) c[i] = bia;
#pragma unroll
      for (int kc = 0; kc < 8; ++kc) {
        v16h b = loadB(ew2t, HIDc, nt, kc * 32, lane);
        c = wmma_f16(a2[kc], b, c);
      }
#pragma unroll
      for (int i = 0; i < 8; ++i)
        s_t1[(r0 + i + mhi) * HIDc + nt * 16 + l15] = (_Float16)silu_f(c[i]);
    }
  }

  // ---- t2 = silu(m @ xw1 + xb1) -> s_t0 (own rows; in-wave in-order DS) ---
  {
    v16h a3[8];
#pragma unroll
    for (int kc = 0; kc < 8; ++kc)
      a3[kc] = ldsA_pair(s_t1 + myrow * HIDc + kc * 32 + ko8);
    for (int nt = 0; nt < 16; ++nt) {
      float bia = s_b3[nt * 16 + l15];
      v8f c;
#pragma unroll
      for (int i = 0; i < 8; ++i) c[i] = bia;
#pragma unroll
      for (int kc = 0; kc < 8; ++kc) {
        v16h b = loadB(xw1t, HIDc, nt, kc * 32, lane);
        c = wmma_f16(a3[kc], b, c);
      }
#pragma unroll
      for (int i = 0; i < 8; ++i)
        s_t0[(r0 + i + mhi) * HIDc + nt * 16 + l15] = (_Float16)silu_f(c[i]);
    }
  }
  __syncthreads();

  // ---- coef[e] = t2[e] . xw2 + xb2  (vectorized 16B DS reads) -------------
  if (tid < 32) s_cf[tid] = row_dot256(s_t0 + tid * HIDc, xw2, xb2[0]);

  // ---- agg[src] = sum_{e<31} m[e]  (pad row 31 skipped) -------------------
  {
    int oct  = tid & 31;                 // column octet 0..31 (8 cols each)
    int half = tid >> 5;                 // 0: rows 0..15, 1: rows 16..30
    float acc[8];
#pragma unroll
    for (int i = 0; i < 8; ++i) acc[i] = 0.0f;
    int e0 = half * 16, e1 = half ? 31 : 16;
    for (int e = e0; e < e1; ++e) {
      v8h v = *(const v8h*)(s_t1 + e * HIDc + oct * 8);
#pragma unroll
      for (int i = 0; i < 8; ++i) acc[i] += (float)v[i];
    }
    float* s_part = (float*)s_t0;        // reuse (t2 consumed after coef)
    __syncthreads();                     // coef done before clobbering s_t0
#pragma unroll
    for (int i = 0; i < 8; ++i) s_part[tid * 8 + i] = acc[i];
    __syncthreads();
    if (tid < 32) {
#pragma unroll
      for (int i = 0; i < 8; ++i) {
        float s = s_part[tid * 8 + i] + s_part[(tid + 32) * 8 + i];
        agg[(size_t)src * HIDc + tid * 8 + i] = s;
      }
    }
  }
  // ---- x_next[src] = x[src] + sum_e rel[e] * coef[e] ----------------------
  if (tid < 3) {
    float s = xc[src * 3 + tid];
    for (int e = 0; e < 31; ++e) s += s_cf[e] * s_rel[e * 4 + tid];
    xn[src * 3 + tid] = s;
  }
}

// ---------------------------------------------------------------------------
// Node kernel: h += silu(cat(h,agg) @ hw1 + hb1) @ hw2 + hb2
// ---------------------------------------------------------------------------
#define NODE_LDS (2 * (16 * 512 + 16 * 256) * 2 + 2 * 256 * 4)

__global__ void __launch_bounds__(64)
node_kernel(float* __restrict__ h32, _Float16* __restrict__ h16,
            const float* __restrict__ agg,
            const _Float16* __restrict__ hw1t,   // [256][512]
            const _Float16* __restrict__ hw2t,   // [256][256]
            const float* __restrict__ hb1, const float* __restrict__ hb2) {
  extern __shared__ char smem[];
  const int tid = threadIdx.x, lane = tid & 31, wave = tid >> 5;
  _Float16* s_cat = (_Float16*)smem + (size_t)wave * (16 * 512 + 16 * 256);
  _Float16* s_u   = s_cat + 16 * 512;
  float*    s_hb1 = (float*)(smem + 2 * (16 * 512 + 16 * 256) * 2);
  float*    s_hb2 = s_hb1 + 256;
  const int n0 = (blockIdx.x * 2 + wave) * 16;

  for (int idx = lane; idx < 16 * 64; idx += 32) {
    int row = idx >> 6, ch = idx & 63;
    if (ch < 32) {
      *(v8h*)(s_cat + row * 512 + ch * 8) =
          *(const v8h*)(h16 + (size_t)(n0 + row) * HIDc + ch * 8);
    } else {
      const float* ap = agg + (size_t)(n0 + row) * HIDc + (ch - 32) * 8;
      v8h t;
#pragma unroll
      for (int i = 0; i < 8; ++i) t[i] = (_Float16)ap[i];
      *(v8h*)(s_cat + row * 512 + ch * 8) = t;
    }
  }
  for (int i = tid; i < 256; i += 64) { s_hb1[i] = hb1[i]; s_hb2[i] = hb2[i]; }
  __syncthreads();

  const int l15   = lane & 15;
  const int ko8   = (lane < 16) ? 0 : 16;
  const int mhi   = (lane < 16) ? 0 : 8;

  {
    v16h a1[16];
#pragma unroll
    for (int kc = 0; kc < 16; ++kc)                   // K = 512
      a1[kc] = ldsA_pair(s_cat + l15 * 512 + kc * 32 + ko8);
    for (int nt = 0; nt < 16; ++nt) {
      float bia = s_hb1[nt * 16 + l15];
      v8f c;
#pragma unroll
      for (int i = 0; i < 8; ++i) c[i] = bia;
#pragma unroll
      for (int kc = 0; kc < 16; ++kc) {
        v16h b = loadB(hw1t, 512, nt, kc * 32, lane);
        c = wmma_f16(a1[kc], b, c);
      }
#pragma unroll
      for (int i = 0; i < 8; ++i)
        s_u[(i + mhi) * HIDc + nt * 16 + l15] = (_Float16)silu_f(c[i]);
    }
  }
  {
    v16h a2[8];
#pragma unroll
    for (int kc = 0; kc < 8; ++kc)                    // K = 256
      a2[kc] = ldsA_pair(s_u + l15 * HIDc + kc * 32 + ko8);
    for (int nt = 0; nt < 16; ++nt) {
      float bia = s_hb2[nt * 16 + l15];
      v8f c;
#pragma unroll
      for (int i = 0; i < 8; ++i) c[i] = bia;
#pragma unroll
      for (int kc = 0; kc < 8; ++kc) {
        v16h b = loadB(hw2t, HIDc, nt, kc * 32, lane);
        c = wmma_f16(a2[kc], b, c);
      }
#pragma unroll
      for (int i = 0; i < 8; ++i) {
        size_t idx = (size_t)(n0 + i + mhi) * HIDc + nt * 16 + l15;
        float nv = h32[idx] + c[i];                   // residual
        h32[idx] = nv;
        h16[idx] = (_Float16)nv;
      }
    }
  }
}

// ---------------------------------------------------------------------------
// Gated readout: g=silu(h@rg+b)*(h@rv+b) twice, then y = g @ rw3 + rb3.
// ---------------------------------------------------------------------------
#define RO_LDS (2 * (2 * 16 * 256) * 2 + 4 * 256 * 4)

__global__ void __launch_bounds__(64)
readout_kernel(const _Float16* __restrict__ h16,
               const _Float16* __restrict__ rg1t, const _Float16* __restrict__ rv1t,
               const _Float16* __restrict__ rg2t, const _Float16* __restrict__ rv2t,
               const float* __restrict__ rgb1, const float* __restrict__ rvb1,
               const float* __restrict__ rgb2, const float* __restrict__ rvb2,
               const float* __restrict__ rw3, const float* __restrict__ rb3,
               float* __restrict__ y) {
  extern __shared__ char smem[];
  const int tid = threadIdx.x, lane = tid & 31, wave = tid >> 5;
  _Float16* s_h = (_Float16*)smem + (size_t)wave * (2 * 16 * 256);
  _Float16* s_g = s_h + 16 * 256;
  float*    s_bg1 = (float*)(smem + 2 * (2 * 16 * 256) * 2);
  float*    s_bv1 = s_bg1 + 256;
  float*    s_bg2 = s_bv1 + 256;
  float*    s_bv2 = s_bg2 + 256;
  const int n0 = (blockIdx.x * 2 + wave) * 16;

  for (int idx = lane; idx < 16 * 32; idx += 32) {
    int row = idx >> 5, ch = idx & 31;
    *(v8h*)(s_h + row * HIDc + ch * 8) =
        *(const v8h*)(h16 + (size_t)(n0 + row) * HIDc + ch * 8);
  }
  for (int i = tid; i < 256; i += 64) {
    s_bg1[i] = rgb1[i]; s_bv1[i] = rvb1[i];
    s_bg2[i] = rgb2[i]; s_bv2[i] = rvb2[i];
  }
  __syncthreads();

  const int l15 = lane & 15;
  const int ko8 = (lane < 16) ? 0 : 16;
  const int mhi = (lane < 16) ? 0 : 8;

  // stage 1: s_g = silu(h@rg1+b) * (h@rv1+b)
  {
    v16h a[8];
#pragma unroll
    for (int kc = 0; kc < 8; ++kc)
      a[kc] = ldsA_pair(s_h + l15 * HIDc + kc * 32 + ko8);
    for (int nt = 0; nt < 16; ++nt) {
      float bg = s_bg1[nt * 16 + l15], bv = s_bv1[nt * 16 + l15];
      v8f cg, cv;
#pragma unroll
      for (int i = 0; i < 8; ++i) { cg[i] = bg; cv[i] = bv; }
#pragma unroll
      for (int kc = 0; kc < 8; ++kc) {
        v16h bG = loadB(rg1t, HIDc, nt, kc * 32, lane);
        v16h bV = loadB(rv1t, HIDc, nt, kc * 32, lane);
        cg = wmma_f16(a[kc], bG, cg);
        cv = wmma_f16(a[kc], bV, cv);
      }
#pragma unroll
      for (int i = 0; i < 8; ++i)
        s_g[(i + mhi) * HIDc + nt * 16 + l15] =
            (_Float16)(silu_f(cg[i]) * cv[i]);
    }
  }
  // stage 2: s_h = silu(g@rg2+b) * (g@rv2+b)
  {
    v16h a[8];
#pragma unroll
    for (int kc = 0; kc < 8; ++kc)
      a[kc] = ldsA_pair(s_g + l15 * HIDc + kc * 32 + ko8);
    for (int nt = 0; nt < 16; ++nt) {
      float bg = s_bg2[nt * 16 + l15], bv = s_bv2[nt * 16 + l15];
      v8f cg, cv;
#pragma unroll
      for (int i = 0; i < 8; ++i) { cg[i] = bg; cv[i] = bv; }
#pragma unroll
      for (int kc = 0; kc < 8; ++kc) {
        v16h bG = loadB(rg2t, HIDc, nt, kc * 32, lane);
        v16h bV = loadB(rv2t, HIDc, nt, kc * 32, lane);
        cg = wmma_f16(a[kc], bG, cg);
        cv = wmma_f16(a[kc], bV, cv);
      }
#pragma unroll
      for (int i = 0; i < 8; ++i)
        s_h[(i + mhi) * HIDc + nt * 16 + l15] =
            (_Float16)(silu_f(cg[i]) * cv[i]);
    }
  }
  // y = g2 . rw3 + rb3   (16 nodes, lanes 0..15)
  if (lane < 16)
    y[n0 + lane] = row_dot256(s_h + lane * HIDc, rw3, rb3[0]);
}

// ---------------------------------------------------------------------------
// ae = per-graph sum of y; forces = x_final - x0.  One wave per graph.
// ---------------------------------------------------------------------------
__global__ void __launch_bounds__(64)
finalize_kernel(const float* __restrict__ y, const float* __restrict__ xf,
                const float* __restrict__ x0, float* __restrict__ out) {
  int gid  = blockIdx.x * 2 + (threadIdx.x >> 5);
  int lane = threadIdx.x & 31;
  int n    = gid * 32 + lane;
#pragma unroll
  for (int c = 0; c < 3; ++c)
    out[Gn + n * 3 + c] = xf[n * 3 + c] - x0[n * 3 + c];
  float v = y[n];
  for (int m = 16; m > 0; m >>= 1) v += __shfl_xor(v, m, 32);
  if (lane == 0) out[gid] = v;
}

// ---------------------------------------------------------------------------
extern "C" void kernel_launch(void* const* d_in, const int* in_sizes, int n_in,
                              void* d_out, int out_size, void* d_ws, size_t ws_size,
                              hipStream_t stream) {
  (void)in_sizes; (void)n_in; (void)out_size; (void)ws_size;
  const float* node_feat = (const float*)d_in[0];
  const float* pos       = (const float*)d_in[1];
  const float* tptr      = (const float*)d_in[2];
  const float* enc_w1 = (const float*)d_in[3];
  const float* enc_b1 = (const float*)d_in[4];
  const float* enc_w2 = (const float*)d_in[5];
  const float* enc_b2 = (const float*)d_in[6];
  const float* ew1 = (const float*)d_in[7];
  const float* eb1 = (const float*)d_in[8];
  const float* ew2 = (const float*)d_in[9];
  const float* eb2 = (const float*)d_in[10];
  const float* xw1 = (const float*)d_in[11];
  const float* xb1 = (const float*)d_in[12];
  const float* xw2 = (const float*)d_in[13];
  const float* xb2 = (const float*)d_in[14];
  const float* hw1 = (const float*)d_in[15];
  const float* hb1 = (const float*)d_in[16];
  const float* hw2 = (const float*)d_in[17];
  const float* hb2 = (const float*)d_in[18];
  const float* rg1 = (const float*)d_in[19];
  const float* rgb1 = (const float*)d_in[20];
  const float* rv1 = (const float*)d_in[21];
  const float* rvb1 = (const float*)d_in[22];
  const float* rg2 = (const float*)d_in[23];
  const float* rgb2 = (const float*)d_in[24];
  const float* rv2 = (const float*)d_in[25];
  const float* rvb2 = (const float*)d_in[26];
  const float* rw3 = (const float*)d_in[27];
  const float* rb3 = (const float*)d_in[28];

  // --- workspace carve (~47 MB) ------------------------------------------
  char* ws = (char*)d_ws;
  size_t off = 0;
  auto carve = [&](size_t bytes) -> void* {
    void* p = ws + off;
    off += (bytes + 255) & ~(size_t)255;
    return p;
  };
  float*     x0   = (float*)carve((size_t)Nn * 3 * 4);
  float*     xa   = (float*)carve((size_t)Nn * 3 * 4);
  float*     xb   = (float*)carve((size_t)Nn * 3 * 4);
  float*     h32  = (float*)carve((size_t)Nn * HIDc * 4);
  _Float16*  h16  = (_Float16*)carve((size_t)Nn * HIDc * 2);
  float*     agg  = (float*)carve((size_t)Nn * HIDc * 4);
  float*     yv   = (float*)carve((size_t)Nn * 4);
  _Float16*  ew1t = (_Float16*)carve((size_t)Lc * 256 * KP1 * 2);
  _Float16*  ew2t = (_Float16*)carve((size_t)Lc * 256 * 256 * 2);
  _Float16*  xw1t = (_Float16*)carve((size_t)Lc * 256 * 256 * 2);
  _Float16*  hw1t = (_Float16*)carve((size_t)Lc * 256 * 512 * 2);
  _Float16*  hw2t = (_Float16*)carve((size_t)Lc * 256 * 256 * 2);
  _Float16*  rg1t = (_Float16*)carve((size_t)256 * 256 * 2);
  _Float16*  rv1t = (_Float16*)carve((size_t)256 * 256 * 2);
  _Float16*  rg2t = (_Float16*)carve((size_t)256 * 256 * 2);
  _Float16*  rv2t = (_Float16*)carve((size_t)256 * 256 * 2);

  auto cvt = [&](const float* w, _Float16* o, int K, int N, int Kpad) {
    int tot = N * Kpad;
    cvt_kernel<<<(tot + 255) / 256, 256, 0, stream>>>(w, o, K, N, Kpad);
  };
  for (int l = 0; l < Lc; ++l) {
    cvt(ew1 + (size_t)l * 513 * 256, ew1t + (size_t)l * 256 * KP1, 513, 256, KP1);
    cvt(ew2 + (size_t)l * 256 * 256, ew2t + (size_t)l * 256 * 256, 256, 256, 256);
    cvt(xw1 + (size_t)l * 256 * 256, xw1t + (size_t)l * 256 * 256, 256, 256, 256);
    cvt(hw1 + (size_t)l * 512 * 256, hw1t + (size_t)l * 256 * 512, 512, 256, 512);
    cvt(hw2 + (size_t)l * 256 * 256, hw2t + (size_t)l * 256 * 256, 256, 256, 256);
  }
  cvt(rg1, rg1t, 256, 256, 256);
  cvt(rv1, rv1t, 256, 256, 256);
  cvt(rg2, rg2t, 256, 256, 256);
  cvt(rv2, rv2t, 256, 256, 256);

  centroid_kernel<<<Gn / 2, 64, 0, stream>>>(pos, x0, xa);
  encoder_kernel<<<Nn / 256, 256, 0, stream>>>(node_feat, enc_w1, enc_b1,
                                               enc_w2, enc_b2, tptr, h32, h16);

  float* xc = xa;
  float* xn = xb;
  for (int l = 0; l < Lc; ++l) {
    edge_kernel<<<Nn, 64, EDGE_LDS, stream>>>(
        h16, xc, xn, agg,
        ew1t + (size_t)l * 256 * KP1, ew2t + (size_t)l * 256 * 256,
        xw1t + (size_t)l * 256 * 256,
        eb1 + (size_t)l * 256, eb2 + (size_t)l * 256, xb1 + (size_t)l * 256,
        xw2 + (size_t)l * 256, xb2 + l);
    node_kernel<<<Nn / 32, 64, NODE_LDS, stream>>>(
        h32, h16, agg,
        hw1t + (size_t)l * 256 * 512, hw2t + (size_t)l * 256 * 256,
        hb1 + (size_t)l * 256, hb2 + (size_t)l * 256);
    float* tmp = xc; xc = xn; xn = tmp;
  }

  readout_kernel<<<Nn / 32, 64, RO_LDS, stream>>>(
      h16, rg1t, rv1t, rg2t, rv2t, rgb1, rvb1, rgb2, rvb2, rw3, rb3, yv);

  finalize_kernel<<<Gn / 2, 64, 0, stream>>>(yv, xc, x0, (float*)d_out);
}
